// SwitchTransformerBlock_15006615733155
// MI455X (gfx1250) — matrix operand order repaired and checked by reading, not compiled, not run on hardware
//
#include <hip/hip_runtime.h>
#include <hip/hip_bf16.h>

typedef _Float16 h16;
typedef __attribute__((ext_vector_type(16))) _Float16 v16h;
typedef __attribute__((ext_vector_type(8)))  float    v8f;

// ---------------------------------------------------------------------------
// Problem constants (from reference): B=4, N=2048, C=512, E=8, H=8, K=2, F=4C
// ---------------------------------------------------------------------------
#define BB   4
#define NN   2048
#define CC   512
#define EE   8
#define HH   8
#define FF   2048
#define TT   (BB * NN)

__device__ __forceinline__ float gelu_exact(float v) {
    return 0.5f * v * (1.0f + erff(v * 0.70710678118654752f));
}

// ---------------------------------------------------------------------------
// 16-lane butterfly reductions via DPP16 (no LDS, no waits).
// Mask set {1,3,7,15}: quad_perm(1,0,3,2)=0xB1, quad_perm(3,2,1,0)=0x1B,
// row_half_mirror=0x141 (xor7), row_mirror=0x140 (xor15).
// Valid butterfly for both max (idempotent) and sum (disjoint partitions).
// ---------------------------------------------------------------------------
template <int CTRL>
__device__ __forceinline__ float dpp_mov_f32(float x) {
    int xi = __builtin_bit_cast(int, x);
    int r  = __builtin_amdgcn_update_dpp(xi, xi, CTRL, 0xF, 0xF, true);
    return __builtin_bit_cast(float, r);
}
__device__ __forceinline__ float redmax16(float x) {
    x = fmaxf(x, dpp_mov_f32<0x0B1>(x));
    x = fmaxf(x, dpp_mov_f32<0x01B>(x));
    x = fmaxf(x, dpp_mov_f32<0x141>(x));
    x = fmaxf(x, dpp_mov_f32<0x140>(x));
    return x;
}
__device__ __forceinline__ float redsum16(float x) {
    x += dpp_mov_f32<0x0B1>(x);
    x += dpp_mov_f32<0x01B>(x);
    x += dpp_mov_f32<0x141>(x);
    x += dpp_mov_f32<0x140>(x);
    return x;
}

// ---------------------------------------------------------------------------
// Weight conversion: fp32 -> f16, once per launch (weights then live in L2).
// n must be a multiple of 1024 (true for all weights here).
// ---------------------------------------------------------------------------
__global__ void wconv_kernel(const float* __restrict__ src, h16* __restrict__ dst) {
    const int i = (blockIdx.x * 256 + threadIdx.x) * 4;
    const float4 v = *(const float4*)(src + i);
    dst[i]     = (h16)v.x;
    dst[i + 1] = (h16)v.y;
    dst[i + 2] = (h16)v.z;
    dst[i + 3] = (h16)v.w;
}

// ---------------------------------------------------------------------------
// LayerNorm: fp32 in -> f16 out.  One block (256 thr) per token, C=512.
// ---------------------------------------------------------------------------
__global__ void ln_kernel(const float* __restrict__ x, const float* __restrict__ w,
                          const float* __restrict__ b, h16* __restrict__ out) {
    const int t = blockIdx.x;
    const float* row = x + (size_t)t * CC;
    const int tid = threadIdx.x;
    float v0 = row[tid], v1 = row[tid + 256];
    float s = v0 + v1, q = v0 * v0 + v1 * v1;
#pragma unroll
    for (int m = 16; m >= 1; m >>= 1) {
        s += __shfl_xor(s, m, 32);
        q += __shfl_xor(q, m, 32);
    }
    __shared__ float ss[8], sq[8];
    __shared__ float mean_s, rstd_s;
    const int wid = tid >> 5;
    if ((tid & 31) == 0) { ss[wid] = s; sq[wid] = q; }
    __syncthreads();
    if (tid == 0) {
        float S = 0.f, Q = 0.f;
#pragma unroll
        for (int i = 0; i < 8; ++i) { S += ss[i]; Q += sq[i]; }
        float mean = S / (float)CC;
        float var  = Q / (float)CC - mean * mean;
        mean_s = mean;
        rstd_s = rsqrtf(var + 1e-5f);
    }
    __syncthreads();
    const float mean = mean_s, rstd = rstd_s;
    out[(size_t)t * CC + tid]       = (h16)((v0 - mean) * rstd * w[tid] + b[tid]);
    out[(size_t)t * CC + tid + 256] = (h16)((v1 - mean) * rstd * w[tid + 256] + b[tid + 256]);
}

// ---------------------------------------------------------------------------
// V transpose: qkv[T,1536] (v part) -> vt[B*512, N] f16  (row = b*512 + h*64+d)
// LDS-tiled 16x16 transpose; grid = (T/16)*(512/16), block 256.
// ---------------------------------------------------------------------------
__global__ void vtrans_kernel(const h16* __restrict__ qkv, h16* __restrict__ vt) {
    __shared__ h16 tile[16][17];
    const int bid = blockIdx.x;
    const int t0 = (bid >> 5) * 16;      // token tile (512 of them)
    const int d0 = (bid & 31) * 16;      // v-column tile (32 of them)
    const int i = threadIdx.x >> 4;
    const int j = threadIdx.x & 15;
    tile[i][j] = qkv[(size_t)(t0 + i) * (3 * CC) + 2 * CC + d0 + j];
    __syncthreads();
    const int b = t0 >> 11;              // batch index
    const int n = t0 & (NN - 1);         // position within batch
    vt[((size_t)b * CC + d0 + i) * NN + n + j] = tile[j][i];
}

// ---------------------------------------------------------------------------
// Gate + top-2 routing.  One wave per token; builds per-expert gather lists.
// imax = max(index of top-2 logits)  (matches reference's last-overwrite MoE).
// ---------------------------------------------------------------------------
__global__ void gate_topk_kernel(const h16* __restrict__ h2,
                                 const float* __restrict__ gw,
                                 const float* __restrict__ gb,
                                 int* __restrict__ counts, int* __restrict__ gather) {
    const int t = blockIdx.x * (blockDim.x >> 5) + (threadIdx.x >> 5);
    if (t >= TT) return;
    const int lane = threadIdx.x & 31;
    float acc[EE];
#pragma unroll
    for (int e = 0; e < EE; ++e) acc[e] = 0.f;
    for (int i = lane; i < CC; i += 32) {
        float hv = (float)h2[(size_t)t * CC + i];
#pragma unroll
        for (int e = 0; e < EE; ++e) acc[e] += hv * gw[e * CC + i];
    }
#pragma unroll
    for (int e = 0; e < EE; ++e)
#pragma unroll
        for (int m = 16; m >= 1; m >>= 1) acc[e] += __shfl_xor(acc[e], m, 32);
    if (lane == 0) {
        float lg[EE];
#pragma unroll
        for (int e = 0; e < EE; ++e) lg[e] = acc[e] + gb[e];
        int i1 = 0;
#pragma unroll
        for (int e = 1; e < EE; ++e) if (lg[e] > lg[i1]) i1 = e;
        int i2 = -1; float best2 = -3.0e38f;
#pragma unroll
        for (int e = 0; e < EE; ++e) {
            if (e == i1) continue;
            if (i2 < 0 || lg[e] > best2) { i2 = e; best2 = lg[e]; }
        }
        const int imax = (i1 > i2) ? i1 : i2;
        const int pos = atomicAdd(&counts[imax], 1);
        gather[imax * TT + pos] = t;
    }
}

// ---------------------------------------------------------------------------
// Generic WMMA GEMM:  Out[M,N] = epi( A[M,K](f16) @ W[N,K]^T(f16) + bias )
// per-wave macro-tile 16(M) x 64(N); K stepped by 32 via v_wmma_f32_16x16x32_f16.
// All loads in the k-loop are contiguous 16B/32B runs -> global_load_b128.
// GATHER: per-expert row lists (gridDim.z = experts).
// ---------------------------------------------------------------------------
template <bool GELU, bool F16OUT, bool HASRES, bool GATHER>
__global__ void wmma_gemm(const h16* __restrict__ A, int lda,
                          const h16* __restrict__ Wm, const float* __restrict__ bias,
                          float* __restrict__ outF, h16* __restrict__ outH, int ldo,
                          const float* __restrict__ res, float res_scale,
                          int M, int N, int K,
                          const int* __restrict__ gather, const int* __restrict__ counts,
                          int tstride) {
    const int e = blockIdx.z;
    const h16*  W  = Wm   + (size_t)e * N * K;
    const float* bv = bias + (size_t)e * N;
    const int count = GATHER ? counts[e] : M;

    const int lane = threadIdx.x & 31;
    const int wid  = threadIdx.x >> 5;
    const int ngroups = N >> 6;
    const int mtiles  = (M + 15) >> 4;
    const int tile = blockIdx.x * (blockDim.x >> 5) + wid;
    if (tile >= mtiles * ngroups) return;
    const int mt = tile / ngroups;
    const int ng = tile % ngroups;
    if (mt * 16 >= count) return;

    const int mcol  = lane & 15;                  // A row / D column within tile
    const int aoff  = (lane < 16) ? 0 : 16;       // A-layout K base (halved)
    const int bhalf = (lane < 16) ? 0 : 16;       // B-layout K base (halved)
    int rowm;
    if (GATHER) {
        int slot = mt * 16 + mcol;
        rowm = gather[e * tstride + (slot < count ? slot : count - 1)];
    } else {
        rowm = mt * 16 + mcol;
    }
    const h16* arow = A + (size_t)rowm * lda + aoff;
    const int ncol0 = ng * 64;

    v8f acc[4] = {};
    for (int k0 = 0; k0 < K; k0 += 32) {
        v16h a;
#pragma unroll
        for (int v = 0; v < 8; ++v) {             // one contiguous 16-half run
            const int k = k0 + 2 * v;
            a[2 * v]     = arow[k];
            a[2 * v + 1] = arow[k + 1];
        }
#pragma unroll
        for (int tn = 0; tn < 4; ++tn) {
            const h16* wb = W + (size_t)(ncol0 + tn * 16 + mcol) * K + k0 + bhalf;
            v16h b;
#pragma unroll
            for (int v = 0; v < 16; ++v) b[v] = wb[v];   // 16 contiguous halfs
            acc[tn] = __builtin_amdgcn_wmma_f32_16x16x32_f16(
                false, a, false, b, (short)0, acc[tn], false, false);
        }
    }

    const int m0 = (lane < 16) ? 0 : 8;
#pragma unroll
    for (int tn = 0; tn < 4; ++tn) {
        const int col = ncol0 + tn * 16 + mcol;
        const float bb = bv[col];
#pragma unroll
        for (int r = 0; r < 8; ++r) {
            const int oslot = mt * 16 + m0 + r;
            if (GATHER && oslot >= count) continue;
            const int orow = GATHER ? gather[e * tstride + oslot] : oslot;
            float v = acc[tn][r] + bb;
            if (GELU)   v = gelu_exact(v);
            if (HASRES) v = res[(size_t)orow * ldo + col] + res_scale * v;
            if (F16OUT) outH[(size_t)orow * ldo + col] = (h16)v;
            else        outF[(size_t)orow * ldo + col] = v;
        }
    }
}

// ---------------------------------------------------------------------------
// Flash attention: one wave per (b, h, 16-row Q tile).  dh=64, scale=1/8.
// qkv: [T,1536] f16 (q|k|v);  vt: [B*512, N] f16 (transposed V).
// DPP16 butterfly reductions; per-wave LDS converts P D-layout -> A-layout.
// __launch_bounds__(128,1): full VGPR budget, no spills.
// Output: attn[T,512] f16.
// ---------------------------------------------------------------------------
__global__ void __launch_bounds__(128, 1)
attn_kernel(const h16* __restrict__ qkv, const h16* __restrict__ vt,
            h16* __restrict__ attn) {
    const int LD = 3 * CC;
    const int ntiles = NN >> 4;                 // 128
    const int lane = threadIdx.x & 31;
    const int wid  = threadIdx.x >> 5;
    int id = blockIdx.x * 4 + wid;              // 0 .. B*H*ntiles-1
    const int mt = id % ntiles; id /= ntiles;
    const int h  = id % HH;
    const int b  = id / HH;

    const size_t tok0 = (size_t)b * NN;
    const h16* Qp = qkv + tok0 * LD + h * 64;
    const h16* Kp = Qp + CC;
    const h16* Vt = vt + ((size_t)b * CC + h * 64) * NN;   // rows = dh, cols = keys

    const int mcol  = lane & 15;
    const int aoff  = (lane < 16) ? 0 : 16;
    const int bhalf = (lane < 16) ? 0 : 16;

    // Q A-fragments (16x64 -> two 16x32 frags), loaded once
    const h16* qrow = Qp + (size_t)(mt * 16 + mcol) * LD + aoff;
    v16h qa0, qa1;
#pragma unroll
    for (int v = 0; v < 8; ++v) {
        const int k = 2 * v;
        qa0[2 * v] = qrow[k];       qa0[2 * v + 1] = qrow[k + 1];
        qa1[2 * v] = qrow[32 + k];  qa1[2 * v + 1] = qrow[32 + k + 1];
    }

    float mrow[8], lrow[8];
#pragma unroll
    for (int r = 0; r < 8; ++r) { mrow[r] = -1.0e30f; lrow[r] = 0.f; }
    v8f o[4] = {};

    __shared__ h16 pls[4][16 * 32];             // per-wave P staging
    h16* P = pls[wid];
    const float scale = 0.125f;                 // 1/sqrt(64)
    const int prow0 = (lane < 16) ? 0 : 8;

    for (int jb = 0; jb < NN; jb += 32) {
        // S = Q K^T for 32 keys (two 16-key N tiles, two K=32 steps each)
        v8f s0 = {}, s1 = {};
        {
            const h16* k0p = Kp + (size_t)(jb + mcol) * LD + bhalf;
            const h16* k1p = Kp + (size_t)(jb + 16 + mcol) * LD + bhalf;
            v16h b00, b01, b10, b11;
#pragma unroll
            for (int v = 0; v < 16; ++v) {      // contiguous 16-half runs
                b00[v] = k0p[v];
                b01[v] = k0p[32 + v];
                b10[v] = k1p[v];
                b11[v] = k1p[32 + v];
            }
            s0 = __builtin_amdgcn_wmma_f32_16x16x32_f16(false, qa0, false, b00, (short)0, s0, false, false);
            s0 = __builtin_amdgcn_wmma_f32_16x16x32_f16(false, qa1, false, b01, (short)0, s0, false, false);
            s1 = __builtin_amdgcn_wmma_f32_16x16x32_f16(false, qa0, false, b10, (short)0, s1, false, false);
            s1 = __builtin_amdgcn_wmma_f32_16x16x32_f16(false, qa1, false, b11, (short)0, s1, false, false);
        }
        // online softmax, one row at a time (short live ranges, DPP reductions)
#pragma unroll
        for (int r = 0; r < 8; ++r) {
            const float rmax = redmax16(fmaxf(s0[r], s1[r]) * scale);
            const float mn   = fmaxf(mrow[r], rmax);
            const float corr = __expf(mrow[r] - mn);
            mrow[r] = mn;
            const float p0 = __expf(s0[r] * scale - mn);
            const float p1 = __expf(s1[r] * scale - mn);
            P[(prow0 + r) * 32 + mcol]      = (h16)p0;
            P[(prow0 + r) * 32 + 16 + mcol] = (h16)p1;
            lrow[r] = lrow[r] * corr + redsum16(p0 + p1);
            o[0][r] *= corr;
            o[1][r] *= corr;
            o[2][r] *= corr;
            o[3][r] *= corr;
        }
        // P (A-layout) from per-wave LDS region
        v16h pa;
        const h16* prow = P + mcol * 32 + aoff;
#pragma unroll
        for (int v = 0; v < 8; ++v) {
            const int k = 2 * v;
            pa[2 * v] = prow[k]; pa[2 * v + 1] = prow[k + 1];
        }
        // O += P @ V  (B-frag from transposed V: 16 contiguous halfs per lane)
#pragma unroll
        for (int tn = 0; tn < 4; ++tn) {
            const h16* vp = Vt + (size_t)(tn * 16 + mcol) * NN + jb + bhalf;
            v16h vb;
#pragma unroll
            for (int v = 0; v < 16; ++v) vb[v] = vp[v];
            o[tn] = __builtin_amdgcn_wmma_f32_16x16x32_f16(false, pa, false, vb, (short)0, o[tn], false, false);
        }
    }

    const int m0 = (lane < 16) ? 0 : 8;
#pragma unroll
    for (int tn = 0; tn < 4; ++tn)
#pragma unroll
        for (int r = 0; r < 8; ++r) {
            const float val = o[tn][r] / lrow[r];
            const size_t t = tok0 + mt * 16 + m0 + r;
            attn[t * CC + h * 64 + tn * 16 + mcol] = (h16)val;
        }
}

// ---------------------------------------------------------------------------
// Host-side launch
// ---------------------------------------------------------------------------
extern "C" void kernel_launch(void* const* d_in, const int* in_sizes, int n_in,
                              void* d_out, int out_size, void* d_ws, size_t ws_size,
                              hipStream_t stream) {
    (void)in_sizes; (void)n_in; (void)out_size; (void)ws_size;
    const float* x    = (const float*)d_in[0];
    const float* ln1w = (const float*)d_in[1];
    const float* ln1b = (const float*)d_in[2];
    const float* Wqkv = (const float*)d_in[3];
    const float* bqkv = (const float*)d_in[4];
    const float* Wout = (const float*)d_in[5];
    const float* bout = (const float*)d_in[6];
    const float* ln2w = (const float*)d_in[7];
    const float* ln2b = (const float*)d_in[8];
    const float* gw   = (const float*)d_in[9];
    const float* gb   = (const float*)d_in[10];
    const float* W1   = (const float*)d_in[11];
    const float* b1   = (const float*)d_in[12];
    const float* W2   = (const float*)d_in[13];
    const float* b2   = (const float*)d_in[14];
    const float* ln3w = (const float*)d_in[15];
    const float* ln3b = (const float*)d_in[16];
    const float* m1w  = (const float*)d_in[17];
    const float* m1b  = (const float*)d_in[18];
    const float* m2w  = (const float*)d_in[19];
    const float* m2b  = (const float*)d_in[20];
    float* out = (float*)d_out;

    char* ws = (char*)d_ws;
    const size_t MiB = 1u << 20;
    h16*   h1    = (h16*)  (ws + 0   * MiB);   //  8 MiB [T,C]   (reused as vt)
    h16*   vt    = h1;                         //  8 MiB [B*512,N] (after qkv GEMM)
    h16*   qkv   = (h16*)  (ws + 8   * MiB);   // 24 MiB [T,3C]
    h16*   attn  = (h16*)  (ws + 32  * MiB);   //  8 MiB [T,C]
    float* x1    = (float*)(ws + 40  * MiB);   // 16 MiB [T,C]
    h16*   h2    = (h16*)  (ws + 56  * MiB);   //  8 MiB [T,C]
    h16*   ubuf  = (h16*)  (ws + 64  * MiB);   // 32 MiB [T,F]
    float* x2    = (float*)(ws + 96  * MiB);   // 16 MiB [T,C]
    h16*   h3    = (h16*)  (ws + 112 * MiB);   //  8 MiB [T,C]
    int*   cnt   = (int*)  (ws + 120 * MiB);   // E ints
    int*   gat   = (int*)  (ws + 120 * MiB + 1024);  // E*T ints
    h16*   Wqkvh = (h16*)  (ws + 121 * MiB);   // 1.5 MiB
    h16*   Wouth = (h16*)  (ws + 123 * MiB);   // 0.5 MiB
    h16*   W1h   = (h16*)  (ws + 124 * MiB);   // 16 MiB
    h16*   W2h   = (h16*)  (ws + 140 * MiB);   // 16 MiB
    h16*   m1h   = (h16*)  (ws + 156 * MiB);   //  2 MiB
    h16*   m2h   = (h16*)  (ws + 158 * MiB);   //  2 MiB

    // --- one-time (per launch) weight downconversion fp32 -> f16 ---
    wconv_kernel<<<(3 * CC * CC) / 1024, 256, 0, stream>>>(Wqkv, Wqkvh);
    wconv_kernel<<<(CC * CC) / 1024,     256, 0, stream>>>(Wout, Wouth);
    wconv_kernel<<<(EE * FF * CC) / 1024,256, 0, stream>>>(W1,   W1h);
    wconv_kernel<<<(EE * CC * FF) / 1024,256, 0, stream>>>(W2,   W2h);
    wconv_kernel<<<(FF * CC) / 1024,     256, 0, stream>>>(m1w,  m1h);
    wconv_kernel<<<(CC * FF) / 1024,     256, 0, stream>>>(m2w,  m2h);

    // --- attention sublayer ---
    ln_kernel<<<TT, 256, 0, stream>>>(x, ln1w, ln1b, h1);
    {   // qkv = h1 @ Wqkv^T + b   -> f16 [T,1536]
        const int tiles = (TT / 16) * ((3 * CC) / 64);
        wmma_gemm<false, true, false, false><<<dim3(tiles / 4), 128, 0, stream>>>(
            h1, CC, Wqkvh, bqkv, nullptr, qkv, 3 * CC, nullptr, 0.f,
            TT, 3 * CC, CC, nullptr, nullptr, 0);
    }
    vtrans_kernel<<<(TT / 16) * (CC / 16), 256, 0, stream>>>(qkv, vt);  // h1 dead now
    attn_kernel<<<dim3((BB * HH * (NN / 16)) / 4), 128, 0, stream>>>(qkv, vt, attn);
    {   // x1 = x + attn @ Wout^T + b
        const int tiles = (TT / 16) * (CC / 64);
        wmma_gemm<false, false, true, false><<<dim3(tiles / 4), 128, 0, stream>>>(
            attn, CC, Wouth, bout, x1, nullptr, CC, x, 1.0f,
            TT, CC, CC, nullptr, nullptr, 0);
    }

    // --- switch MoE sublayer ---
    ln_kernel<<<TT, 256, 0, stream>>>(x1, ln2w, ln2b, h2);
    hipMemsetAsync(cnt, 0, EE * sizeof(int), stream);
    gate_topk_kernel<<<TT / 8, 256, 0, stream>>>(h2, gw, gb, cnt, gat);
    {   // u = gelu(h2 @ W1[e]^T + b1[e])  (gathered rows, per expert)
        const int tiles = (TT / 16) * (FF / 64);
        wmma_gemm<true, true, false, true><<<dim3(tiles / 4, 1, EE), 128, 0, stream>>>(
            h2, CC, W1h, b1, nullptr, ubuf, FF, nullptr, 0.f,
            TT, FF, CC, gat, cnt, TT);
    }
    {   // x2 = x1 + top_k * (u @ W2[e]^T + b2[e])
        const int tiles = (TT / 16) * (CC / 64);
        wmma_gemm<false, false, true, true><<<dim3(tiles / 4, 1, EE), 128, 0, stream>>>(
            ubuf, FF, W2h, b2, x2, nullptr, CC, x1, 2.0f,
            TT, CC, FF, gat, cnt, TT);
    }

    // --- dense MLP sublayer ---
    ln_kernel<<<TT, 256, 0, stream>>>(x2, ln3w, ln3b, h3);
    {   // u = gelu(h3 @ m1^T + b)
        const int tiles = (TT / 16) * (FF / 64);
        wmma_gemm<true, true, false, false><<<dim3(tiles / 4), 128, 0, stream>>>(
            h3, CC, m1h, m1b, nullptr, ubuf, FF, nullptr, 0.f,
            TT, FF, CC, nullptr, nullptr, 0);
    }
    {   // out = x2 + u @ m2^T + b
        const int tiles = (TT / 16) * (CC / 64);
        wmma_gemm<false, false, true, false><<<dim3(tiles / 4), 128, 0, stream>>>(
            ubuf, FF, m2h, m2b, out, nullptr, CC, x2, 1.0f,
            TT, CC, FF, nullptr, nullptr, 0);
    }
}
